// TitansMACMemory_67405216744174
// MI455X (gfx1250) — matrix operand order repaired and checked by reading, not compiled, not run on hardware
//
#include <hip/hip_runtime.h>
#include <hip/hip_bf16.h>

#define Bdim 8192
#define Hdim 4096
#define Mdim 64
#define BETA_C  0.9f
#define ALPHA_C 0.99f
#define CGRAD   (2.0f / 33554432.0f)   // 2 / (B*H)

typedef __attribute__((ext_vector_type(16))) __bf16 bf16x16;
typedef __attribute__((ext_vector_type(8)))  __bf16 bf16x8;
typedef __attribute__((ext_vector_type(8)))  float  f32x8;
typedef __attribute__((ext_vector_type(4)))  float  f32x4;

__device__ __forceinline__ bf16x16 pack16(bf16x8 lo, bf16x8 hi) {
  bf16x16 r;
#pragma unroll
  for (int i = 0; i < 8; i++) { r[i] = lo[i]; r[i + 8] = hi[i]; }
  return r;
}

// A-operand (16x32, MxK): lane holds row M=lane%16; 8 halves at k0=(lane>=16?8:0), 8 at k0+16
__device__ __forceinline__ bf16x16 load_a_f32(const float* __restrict__ row, int kbase, int lane) {
  int k0 = kbase + ((lane & 16) ? 8 : 0);
  f32x4 a0 = *(const f32x4*)(row + k0);
  f32x4 a1 = *(const f32x4*)(row + k0 + 4);
  f32x4 a2 = *(const f32x4*)(row + k0 + 16);
  f32x4 a3 = *(const f32x4*)(row + k0 + 20);
  bf16x16 r;
#pragma unroll
  for (int i = 0; i < 4; i++) {
    r[i]      = (__bf16)a0[i];
    r[i + 4]  = (__bf16)a1[i];
    r[i + 8]  = (__bf16)a2[i];
    r[i + 12] = (__bf16)a3[i];
  }
  return r;
}

__device__ __forceinline__ bf16x16 load_a_bf16(const __bf16* __restrict__ row, int kbase, int lane) {
  int k0 = kbase + ((lane & 16) ? 8 : 0);
  return pack16(*(const bf16x8*)(row + k0), *(const bf16x8*)(row + k0 + 16));
}

// B-operand (32x16, KxN): lane holds col N=lane%16; 8 K at k0=(lane>=16?8:0), 8 at k0+16
__device__ __forceinline__ bf16x16 load_b_bf16(const __bf16* __restrict__ row, int kbase, int lane) {
  int k0 = kbase + ((lane & 16) ? 8 : 0);
  return pack16(*(const bf16x8*)(row + k0), *(const bf16x8*)(row + k0 + 16));
}

__device__ __forceinline__ f32x8 wmma_bf(bf16x16 a, bf16x16 b, f32x8 c) {
  return __builtin_amdgcn_wmma_f32_16x16x32_bf16(false, a, false, b, (short)0, c, false, false);
}

__device__ __forceinline__ float gelu_f(float x) {
  return 0.5f * x * (1.0f + erff(x * 0.70710678118654752f));
}
__device__ __forceinline__ float dgelu_f(float x) {
  float cdf = 0.5f * (1.0f + erff(x * 0.70710678118654752f));
  float pdf = 0.3989422804014327f * __expf(-0.5f * x * x);
  return cdf + x * pdf;
}

// ---------- K0: bf16 copies of W1 (M,H) and W2^T (M,H) ----------
__global__ __launch_bounds__(256) void k_prep(const float* __restrict__ W1,
                                              const float* __restrict__ W2,
                                              __bf16* __restrict__ W1bf,
                                              __bf16* __restrict__ W2Tbf) {
  int i = blockIdx.x * 256 + threadIdx.x;           // over M*H
  int m = i / Hdim, h = i % Hdim;
  W1bf[i]  = (__bf16)W1[i];
  W2Tbf[i] = (__bf16)W2[(size_t)h * Mdim + m];
}

// ---------- K1: A = X@W1^T  (store At f32, Gt=gelu bf16, both (M,B))
//                XW2 = X@W2 via W2T rows (store XW2t f32 (M,B)) ----------
__global__ __launch_bounds__(256) void k_fwd1(const float* __restrict__ X,
                                              const __bf16* __restrict__ W1bf,
                                              const __bf16* __restrict__ W2Tbf,
                                              float* __restrict__ At,
                                              float* __restrict__ XW2t,
                                              __bf16* __restrict__ Gt) {
  int tid = threadIdx.x, wave = tid >> 5, lane = tid & 31;
  int rb = blockIdx.x * 16;
  bool isA = wave < 4;
  int mt = (isA ? wave : wave - 4) * 16;
  const float*  xrow = X + (size_t)(rb + (lane & 15)) * Hdim;
  const __bf16* wrow = (isA ? W1bf : W2Tbf) + (size_t)(mt + (lane & 15)) * Hdim;
  f32x8 acc = {};
  for (int k = 0; k < Hdim; k += 32)
    acc = wmma_bf(load_a_f32(xrow, k, lane), load_b_bf16(wrow, k, lane), acc);

  int m  = mt + (lane & 15);                 // output column index (weight row)
  int b0 = rb + ((lane & 16) ? 8 : 0);       // output row base (token row)
  if (isA) {
    f32x4 lo = {acc[0], acc[1], acc[2], acc[3]};
    f32x4 hi = {acc[4], acc[5], acc[6], acc[7]};
    *(f32x4*)(At + (size_t)m * Bdim + b0)     = lo;
    *(f32x4*)(At + (size_t)m * Bdim + b0 + 4) = hi;
    bf16x8 g;
#pragma unroll
    for (int v = 0; v < 8; v++) g[v] = (__bf16)gelu_f(acc[v]);
    *(bf16x8*)(Gt + (size_t)m * Bdim + b0) = g;
  } else {
    f32x4 lo = {acc[0], acc[1], acc[2], acc[3]};
    f32x4 hi = {acc[4], acc[5], acc[6], acc[7]};
    *(f32x4*)(XW2t + (size_t)m * Bdim + b0)     = lo;
    *(f32x4*)(XW2t + (size_t)m * Bdim + b0 + 4) = hi;
  }
}

// ---------- K2: S3 = W2^T W2 (from W2T rows, K=H); S2 = G^T G (from Gt rows, K=B) ----------
__global__ __launch_bounds__(1024) void k_gram(const __bf16* __restrict__ W2Tbf,
                                               const __bf16* __restrict__ Gt,
                                               float* __restrict__ S3,
                                               float* __restrict__ S2) {
  int tid = threadIdx.x, wave = tid >> 5, lane = tid & 31;
  bool first = wave < 16;
  int w  = first ? wave : wave - 16;
  int it = (w >> 2) * 16, jt = (w & 3) * 16;
  const __bf16* base = first ? W2Tbf : Gt;
  int K = first ? Hdim : Bdim;
  const __bf16* arow = base + (size_t)(it + (lane & 15)) * K;
  const __bf16* brow = base + (size_t)(jt + (lane & 15)) * K;
  f32x8 acc = {};
  for (int k = 0; k < K; k += 32)
    acc = wmma_bf(load_a_bf16(arow, k, lane), load_b_bf16(brow, k, lane), acc);
  float* S = first ? S3 : S2;
  int j  = jt + (lane & 15);
  int i0 = it + ((lane & 16) ? 8 : 0);
#pragma unroll
  for (int v = 0; v < 8; v++) S[(i0 + v) * Mdim + j] = acc[v];
}

// ---------- K3: dAt[n,b] = CGRAD*(sum_m Gt[m,b]*S3[m,n] - XW2t[n,b]) * gelu'(At[n,b]) ----------
__global__ __launch_bounds__(256) void k_dA(const __bf16* __restrict__ Gt,
                                            const float* __restrict__ XW2t,
                                            const float* __restrict__ At,
                                            const float* __restrict__ S3,
                                            __bf16* __restrict__ dAt) {
  int b = blockIdx.x * 256 + threadIdx.x;
  int n = blockIdx.y;
  float acc = 0.f;
#pragma unroll 8
  for (int m = 0; m < Mdim; m++)
    acc = fmaf((float)Gt[(size_t)m * Bdim + b], S3[m * Mdim + n], acc);
  float dG = CGRAD * (acc - XW2t[(size_t)n * Bdim + b]);
  float dA = dG * dgelu_f(At[(size_t)n * Bdim + b]);
  dAt[(size_t)n * Bdim + b] = (__bf16)dA;
}

// ---------- K4: g1 = dA^T X (K=B, X cols staged via LDS), fused -> W1n bf16 (M,H) ----------
__global__ __launch_bounds__(128) void k_g1(const float* __restrict__ X,
                                            const __bf16* __restrict__ dAt,
                                            const float* __restrict__ W1,
                                            const float* __restrict__ m1,
                                            const float* __restrict__ log_eta,
                                            const float* __restrict__ gate,
                                            __bf16* __restrict__ W1nbf) {
  __shared__ __bf16 xT[16 * 40];                    // [h_local][b_local], padded stride 40
  int tid = threadIdx.x, wave = tid >> 5, lane = tid & 31;
  int h0 = blockIdx.x * 16;
  const __bf16* arow = dAt + (size_t)(wave * 16 + (lane & 15)) * Bdim;
  int bl = tid >> 2, hq = (tid & 3) * 4;
  f32x8 acc = {};
  for (int b0 = 0; b0 < Bdim; b0 += 32) {
    f32x4 xv = *(const f32x4*)(X + (size_t)(b0 + bl) * Hdim + h0 + hq);
    __syncthreads();
#pragma unroll
    for (int j = 0; j < 4; j++) xT[(hq + j) * 40 + bl] = (__bf16)xv[j];
    __syncthreads();
    int k0 = (lane & 16) ? 8 : 0;
    const __bf16* xr = &xT[(lane & 15) * 40 + k0];
    bf16x16 bop = pack16(*(const bf16x8*)xr, *(const bf16x8*)(xr + 16));
    acc = wmma_bf(load_a_bf16(arow, b0, lane), bop, acc);
  }
  float ec = __expf(log_eta[0]) * gate[0];
  int h  = h0 + (lane & 15);
  int m0 = wave * 16 + ((lane & 16) ? 8 : 0);
#pragma unroll
  for (int v = 0; v < 8; v++) {
    int m = m0 + v;
    float g1v = CGRAD * acc[v];
    float wv = ALPHA_C * W1[(size_t)m * Hdim + h] + BETA_C * m1[(size_t)m * Hdim + h] - ec * g1v;
    W1nbf[(size_t)m * Hdim + h] = (__bf16)wv;
  }
}

// ---------- K5: Y = G^T X (K=B), g2[h,n] = CGRAD*((W2*S2)[h,n] - Y[n,h]),
//                fused -> W2n bf16 (H,M) ----------
__global__ __launch_bounds__(128) void k_g2(const float* __restrict__ X,
                                            const __bf16* __restrict__ Gt,
                                            const float* __restrict__ W2,
                                            const float* __restrict__ m2,
                                            const float* __restrict__ S2,
                                            const float* __restrict__ log_eta,
                                            const float* __restrict__ gate,
                                            __bf16* __restrict__ W2nbf) {
  __shared__ __bf16 xT[16 * 40];
  int tid = threadIdx.x, wave = tid >> 5, lane = tid & 31;
  int h0 = blockIdx.x * 16;
  const __bf16* arow = Gt + (size_t)(wave * 16 + (lane & 15)) * Bdim;
  int bl = tid >> 2, hq = (tid & 3) * 4;
  f32x8 acc = {};
  for (int b0 = 0; b0 < Bdim; b0 += 32) {
    f32x4 xv = *(const f32x4*)(X + (size_t)(b0 + bl) * Hdim + h0 + hq);
    __syncthreads();
#pragma unroll
    for (int j = 0; j < 4; j++) xT[(hq + j) * 40 + bl] = (__bf16)xv[j];
    __syncthreads();
    int k0 = (lane & 16) ? 8 : 0;
    const __bf16* xr = &xT[(lane & 15) * 40 + k0];
    bf16x16 bop = pack16(*(const bf16x8*)xr, *(const bf16x8*)(xr + 16));
    acc = wmma_bf(load_a_bf16(arow, b0, lane), bop, acc);
  }
  float ec = __expf(log_eta[0]) * gate[0];
  int h  = h0 + (lane & 15);
  int n0 = wave * 16 + ((lane & 16) ? 8 : 0);
  const float* w2r = W2 + (size_t)h * Mdim;
  const float* m2r = m2 + (size_t)h * Mdim;
  bf16x8 wout;
#pragma unroll
  for (int v = 0; v < 8; v++) {
    int n = n0 + v;
    const float* s2r = S2 + n * Mdim;               // S2 symmetric
    float wsum = 0.f;
#pragma unroll 8
    for (int mp = 0; mp < Mdim; mp++) wsum = fmaf(w2r[mp], s2r[mp], wsum);
    float g2v = CGRAD * (wsum - acc[v]);
    float wv  = ALPHA_C * w2r[n] + BETA_C * m2r[n] - ec * g2v;
    wout[v] = (__bf16)wv;
  }
  *(bf16x8*)(W2nbf + (size_t)h * Mdim + n0) = wout;
}

// ---------- K6: out = gelu(X@W1n^T) @ W2n^T, gelu fused through LDS ----------
__global__ __launch_bounds__(256) void k_fwd2(const float* __restrict__ X,
                                              const __bf16* __restrict__ W1nbf,
                                              const __bf16* __restrict__ W2nbf,
                                              float* __restrict__ out) {
  __shared__ __bf16 g2s[16 * 72];                   // gelu(A2) block, padded stride 72
  __shared__ float partial[16 * 65];                // K-split partial sums
  int tid = threadIdx.x, wave = tid >> 5, lane = tid & 31;
  int rb = blockIdx.x * 16;
  // Phase A: 8 waves = 4 m-tiles x 2 K-halves
  int w4 = wave & 3;
  const float*  xrow = X + (size_t)(rb + (lane & 15)) * Hdim;
  const __bf16* wrow = W1nbf + (size_t)(w4 * 16 + (lane & 15)) * Hdim;
  int kbeg = (wave & 4) ? (Hdim / 2) : 0;
  f32x8 acc = {};
  for (int k = kbeg; k < kbeg + Hdim / 2; k += 32)
    acc = wmma_bf(load_a_f32(xrow, k, lane), load_b_bf16(wrow, k, lane), acc);
  int m   = w4 * 16 + (lane & 15);
  int bl0 = (lane & 16) ? 8 : 0;
  if (wave & 4) {
#pragma unroll
    for (int v = 0; v < 8; v++) partial[(bl0 + v) * 65 + m] = acc[v];
  }
  __syncthreads();
  if (!(wave & 4)) {
#pragma unroll
    for (int v = 0; v < 8; v++)
      g2s[(bl0 + v) * 72 + m] = (__bf16)gelu_f(acc[v] + partial[(bl0 + v) * 65 + m]);
  }
  __syncthreads();
  // Phase B: out block (16 x H), K=64; A-operand loaded once from LDS, reused for 32 tiles/wave
  int k0 = (lane & 16) ? 8 : 0;
  const __bf16* gr = &g2s[(lane & 15) * 72];
  bf16x16 a0 = pack16(*(const bf16x8*)(gr + k0),      *(const bf16x8*)(gr + k0 + 16));
  bf16x16 a1 = pack16(*(const bf16x8*)(gr + 32 + k0), *(const bf16x8*)(gr + 32 + k0 + 16));
#pragma unroll 2
  for (int t = 0; t < 32; t++) {
    int hn = (wave * 32 + t) * 16;
    const __bf16* brow = W2nbf + (size_t)(hn + (lane & 15)) * Mdim;
    f32x8 o = {};
    o = wmma_bf(a0, load_b_bf16(brow, 0, lane), o);
    o = wmma_bf(a1, load_b_bf16(brow, 32, lane), o);
    int h = hn + (lane & 15);
#pragma unroll
    for (int v = 0; v < 8; v++) out[(size_t)(rb + bl0 + v) * Hdim + h] = o[v];
  }
}

extern "C" void kernel_launch(void* const* d_in, const int* in_sizes, int n_in,
                              void* d_out, int out_size, void* d_ws, size_t ws_size,
                              hipStream_t stream) {
  const float* X       = (const float*)d_in[0];
  const float* W1      = (const float*)d_in[1];
  const float* W2      = (const float*)d_in[2];
  const float* m1      = (const float*)d_in[3];
  const float* m2      = (const float*)d_in[4];
  const float* log_eta = (const float*)d_in[5];
  const float* gate    = (const float*)d_in[6];
  float* out = (float*)d_out;

  char* ws = (char*)d_ws;
  float*  At    = (float*)(ws + 0);                              // 2 MB (M,B) f32
  float*  XW2t  = (float*)(ws + (2u << 20));                     // 2 MB (M,B) f32
  __bf16* Gt    = (__bf16*)(ws + (4u << 20));                    // 1 MB (M,B) bf16
  __bf16* dAt   = (__bf16*)(ws + (5u << 20));                    // 1 MB (M,B) bf16
  __bf16* W1bf  = (__bf16*)(ws + (6u << 20));                    // 512 KB (M,H)
  __bf16* W2Tbf = (__bf16*)(ws + (6u << 20) + (512u << 10));     // 512 KB (M,H)
  __bf16* W1nbf = (__bf16*)(ws + (7u << 20));                    // 512 KB (M,H)
  __bf16* W2nbf = (__bf16*)(ws + (7u << 20) + (512u << 10));     // 512 KB (H,M)
  float*  S3    = (float*)(ws + (8u << 20));                     // 16 KB  (W2^T W2)
  float*  S2    = (float*)(ws + (8u << 20) + (16u << 10));       // 16 KB  (G^T G)

  k_prep<<<(Mdim * Hdim) / 256, 256, 0, stream>>>(W1, W2, W1bf, W2Tbf);
  k_fwd1<<<Bdim / 16, 256, 0, stream>>>(X, W1bf, W2Tbf, At, XW2t, Gt);
  k_gram<<<1, 1024, 0, stream>>>(W2Tbf, Gt, S3, S2);
  dim3 gdA(Bdim / 256, Mdim);
  k_dA<<<gdA, 256, 0, stream>>>(Gt, XW2t, At, S3, dAt);
  k_g1<<<Hdim / 16, 128, 0, stream>>>(X, dAt, W1, m1, log_eta, gate, W1nbf);
  k_g2<<<Hdim / 16, 128, 0, stream>>>(X, Gt, W2, m2, S2, log_eta, gate, W2nbf);
  k_fwd2<<<Bdim / 16, 256, 0, stream>>>(X, W1nbf, W2nbf, out);
}
